// D2Net_78082505441412
// MI455X (gfx1250) — matrix operand order repaired and checked by reading, not compiled, not run on hardware
//
#include <hip/hip_runtime.h>
#include <hip/hip_bf16.h>

typedef __bf16 bf16;
typedef __attribute__((ext_vector_type(16))) __bf16 v16bf;
typedef __attribute__((ext_vector_type(8)))  __bf16 v8bf;
typedef __attribute__((ext_vector_type(8)))  float  v8f;

// ---------------------------------------------------------------------------
// Layer 1: direct 3->64 conv (K=27 too small for WMMA), fp32, ReLU
// ---------------------------------------------------------------------------
__global__ void conv3x3_direct_c3(const float* __restrict__ in,
                                  const float* __restrict__ w,
                                  const float* __restrict__ bias,
                                  float* __restrict__ out,
                                  int H, int W, int Cout) {
  int gid = blockIdx.x * blockDim.x + threadIdx.x;
  int total = Cout * H * W;
  if (gid >= total) return;
  int x = gid % W;
  int y = (gid / W) % H;
  int co = gid / (W * H);
  float acc = bias[co];
  for (int ci = 0; ci < 3; ++ci) {
    for (int r = 0; r < 3; ++r) {
      int yy = y + r - 1;
      if (yy < 0 || yy >= H) continue;
      const float* row = in + (ci * H + yy) * W;
      const float* wr  = w + ((co * 3 + ci) * 3 + r) * 3;
      for (int s = 0; s < 3; ++s) {
        int xx = x + s - 1;
        if (xx < 0 || xx >= W) continue;
        acc = fmaf(row[xx], wr[s], acc);
      }
    }
  }
  out[gid] = fmaxf(acc, 0.f);
}

// ---------------------------------------------------------------------------
// Repack fp32 weights (Cout,Cin,3,3) -> bf16 tap-major [tap][co][ci]
// ---------------------------------------------------------------------------
__global__ void repack_w_bf16(const float* __restrict__ w, bf16* __restrict__ wt,
                              int Cout, int Cin) {
  int gid = blockIdx.x * blockDim.x + threadIdx.x;
  int total = Cout * Cin * 9;
  if (gid >= total) return;
  int tap = gid % 9;
  int ci  = (gid / 9) % Cin;
  int co  = gid / (9 * Cin);
  wt[((size_t)tap * Cout + co) * Cin + ci] = (bf16)w[gid];
}

// ---------------------------------------------------------------------------
// Implicit-GEMM conv via v_wmma_f32_16x16x32_bf16, halo-staged.
// Block tile: (64*MR) out-channels x 64 contiguous x-pixels of one output row.
// Wave tile: MR co-rows x 2 px-cols = 2*MR accumulators -> A and B fragments
// reused 2x and MR times respectively (MR=2: 8 b128 LDS loads per 4 WMMAs).
// Per k-chunk (32 input channels): stage all 9 taps' weights + a 3-row input
// halo once, then run the fully-unrolled 9-tap WMMA loop from LDS.
// ---------------------------------------------------------------------------
template <int MR>
__launch_bounds__(256)
__global__ void conv3x3_wmma(const float* __restrict__ in,
                             const bf16*  __restrict__ wt,
                             const float* __restrict__ bias,
                             float* __restrict__ out,
                             int Cin, int Cout, int H, int W, int dil) {
  constexpr int CTILE = 64 * MR;
  __shared__ __align__(16) bf16 Alds[9 * CTILE * 32];  // [tap][co][ci]
  __shared__ __align__(16) bf16 Bh[3 * 68 * 32];       // [row][px][ci]  13 KB

  const int t    = threadIdx.x;
  const int wave = t >> 5;
  const int lane = t & 31;
  const int x0 = blockIdx.x * 64;
  const int y  = blockIdx.y;
  const int cb = blockIdx.z * CTILE;       // output-channel base
  const int HW = H * W;
  const int P  = 64 + 2 * dil;             // halo width (<= 68)

  v8f acc[MR][2];
  #pragma unroll
  for (int mr = 0; mr < MR; ++mr) { acc[mr][0] = (v8f){}; acc[mr][1] = (v8f){}; }
  const int mrow0 = (wave >> 1) * MR;      // first 16-co sub-row of this wave
  const int nbase = (wave & 1) * 32;       // pixel sub-base (two 16-px tiles)

  const int kchunks = Cin >> 5;

  for (int kc = 0; kc < kchunks; ++kc) {
    __syncthreads();
    // ---- stage weights for all 9 taps: Wt[tap][cb+co][kc*32 + ci8..] ----
    for (int task = t; task < CTILE * 4; task += 256) {
      const int wco  = task >> 2;
      const int wci8 = (task & 3) << 3;
      const bf16* srcb = wt + ((size_t)(cb + wco)) * Cin + (kc << 5) + wci8;
      const size_t tapstride = (size_t)Cout * Cin;
      #pragma unroll
      for (int tap = 0; tap < 9; ++tap)
        *(v8bf*)&Alds[(tap * CTILE + wco) * 32 + wci8] =
            *(const v8bf*)(srcb + (size_t)tap * tapstride);
      if (kc + 1 < kchunks) __builtin_prefetch(srcb + 32, 0, 0);
    }
    // ---- stage input halo: 3 rows x P px x 32 ci, fp32 -> packed bf16 ----
    // tasks: row(3) x ci-pair(16) x 8px-group(9); groups indexed on 8-aligned
    // global x (x0 - 8 + 8g) so interior loads are aligned float4 pairs.
    for (int task = t; task < 3 * 16 * 9; task += 256) {
      const int g   = task % 9;
      const int ci2 = (task / 9) % 16;
      const int row = task / (9 * 16);
      const int ci  = ci2 * 2;
      const int gx0 = x0 - 8 + g * 8;                 // 8-aligned global x base
      const int px0 = g * 8 - 8 + dil;                // halo px of first element
      const int yy  = y + (row - 1) * dil;
      const bool yok = (yy >= 0) && (yy < H);
      const float* s0 = in + (size_t)((kc << 5) + ci) * HW + (size_t)(yok ? yy : 0) * W + gx0;
      const float* s1 = s0 + HW;
      if (yok && gx0 >= 0 && (gx0 + 7) < W && px0 >= 0 && (px0 + 7) < P) {
        float4 a0 = *(const float4*)s0;
        float4 a1 = *(const float4*)(s0 + 4);
        float4 b0 = *(const float4*)s1;
        float4 b1 = *(const float4*)(s1 + 4);
        unsigned* dst = (unsigned*)&Bh[(row * 68 + px0) * 32 + ci];
        union { bf16 h[2]; unsigned u; } pk;
        pk.h[0] = (bf16)a0.x; pk.h[1] = (bf16)b0.x; dst[0 * 16] = pk.u;
        pk.h[0] = (bf16)a0.y; pk.h[1] = (bf16)b0.y; dst[1 * 16] = pk.u;
        pk.h[0] = (bf16)a0.z; pk.h[1] = (bf16)b0.z; dst[2 * 16] = pk.u;
        pk.h[0] = (bf16)a0.w; pk.h[1] = (bf16)b0.w; dst[3 * 16] = pk.u;
        pk.h[0] = (bf16)a1.x; pk.h[1] = (bf16)b1.x; dst[4 * 16] = pk.u;
        pk.h[0] = (bf16)a1.y; pk.h[1] = (bf16)b1.y; dst[5 * 16] = pk.u;
        pk.h[0] = (bf16)a1.z; pk.h[1] = (bf16)b1.z; dst[6 * 16] = pk.u;
        pk.h[0] = (bf16)a1.w; pk.h[1] = (bf16)b1.w; dst[7 * 16] = pk.u;
      } else {
        #pragma unroll
        for (int j = 0; j < 8; ++j) {
          const int px = px0 + j;
          if (px < 0 || px >= P) continue;
          const int xg = gx0 + j;
          const bool ok = yok && (xg >= 0) && (xg < W);
          union { bf16 h[2]; unsigned u; } pk;
          pk.h[0] = (bf16)(ok ? s0[j] : 0.f);
          pk.h[1] = (bf16)(ok ? s1[j] : 0.f);
          *(unsigned*)&Bh[(row * 68 + px) * 32 + ci] = pk.u;
        }
      }
    }
    __syncthreads();

    // ---- 9-tap compute, fragments straight from LDS ----
    const int m    = lane & 15;
    const int ksel = lane >> 4;
    const int kbA  = ksel * 8;
    const int kbB  = ksel * 8;
    #pragma unroll
    for (int tap = 0; tap < 9; ++tap) {
      const int r = tap / 3, s = tap % 3;
      v16bf a[MR];
      #pragma unroll
      for (int mr = 0; mr < MR; ++mr) {
        const bf16* ap = &Alds[(tap * CTILE + (mrow0 + mr) * 16 + m) * 32 + kbA];
        v8bf lo = *(const v8bf*)ap;
        v8bf hi = *(const v8bf*)(ap + 16);
        #pragma unroll
        for (int i = 0; i < 8; ++i) { a[mr][i] = lo[i]; a[mr][i + 8] = hi[i]; }
      }
      const int pxa = nbase + m + s * dil;    // halo px for this output column
      v16bf b0, b1;
      {
        const bf16* bp = &Bh[(r * 68 + pxa) * 32 + kbB];
        v8bf lo = *(const v8bf*)bp;
        v8bf hi = *(const v8bf*)(bp + 16);
        #pragma unroll
        for (int i = 0; i < 8; ++i) { b0[i] = lo[i]; b0[i + 8] = hi[i]; }
      }
      {
        const bf16* bp = &Bh[(r * 68 + pxa + 16) * 32 + kbB];
        v8bf lo = *(const v8bf*)bp;
        v8bf hi = *(const v8bf*)(bp + 16);
        #pragma unroll
        for (int i = 0; i < 8; ++i) { b1[i] = lo[i]; b1[i + 8] = hi[i]; }
      }
      #pragma unroll
      for (int mr = 0; mr < MR; ++mr) {
        acc[mr][0] = __builtin_amdgcn_wmma_f32_16x16x32_bf16(
            false, a[mr], false, b0, (short)0, acc[mr][0], false, false);
        acc[mr][1] = __builtin_amdgcn_wmma_f32_16x16x32_bf16(
            false, a[mr], false, b1, (short)0, acc[mr][1], false, false);
      }
    }
  }
  // ---- epilogue: bias + ReLU, fp32 store (C/D layout: lane=N, vgpr=M) ----
  const int m    = lane & 15;
  const int half = lane >> 4;
  #pragma unroll
  for (int mr = 0; mr < MR; ++mr) {
    #pragma unroll
    for (int v = 0; v < 8; ++v) {
      const int co = cb + (mrow0 + mr) * 16 + half * 8 + v;
      const float bb = bias[co];
      int x = x0 + nbase + m;
      if (x < W) out[(size_t)co * HW + y * W + x] = fmaxf(acc[mr][0][v] + bb, 0.f);
      x = x0 + nbase + 16 + m;
      if (x < W) out[(size_t)co * HW + y * W + x] = fmaxf(acc[mr][1][v] + bb, 0.f);
    }
  }
}

// ---------------------------------------------------------------------------
// Pools
// ---------------------------------------------------------------------------
__global__ void maxpool2(const float* __restrict__ in, float* __restrict__ out,
                         int C, int H, int W) {
  int Ho = H >> 1, Wo = W >> 1;
  int gid = blockIdx.x * blockDim.x + threadIdx.x;
  int total = C * Ho * Wo;
  if (gid >= total) return;
  int x = gid % Wo, y = (gid / Wo) % Ho, c = gid / (Wo * Ho);
  const float* p = in + ((size_t)c * H + 2 * y) * W + 2 * x;
  out[gid] = fmaxf(fmaxf(p[0], p[1]), fmaxf(p[W], p[W + 1]));
}

__global__ void avgpool2_s1(const float* __restrict__ in, float* __restrict__ out,
                            int C, int H, int W) {
  int Ho = H - 1, Wo = W - 1;
  int gid = blockIdx.x * blockDim.x + threadIdx.x;
  int total = C * Ho * Wo;
  if (gid >= total) return;
  int x = gid % Wo, y = (gid / Wo) % Ho, c = gid / (Wo * Ho);
  const float* p = in + ((size_t)c * H + y) * W + x;
  out[gid] = 0.25f * (p[0] + p[1] + p[W] + p[W + 1]);
}

// ---------------------------------------------------------------------------
// Heads
// ---------------------------------------------------------------------------
__global__ void chan_max(const float* __restrict__ f, float* __restrict__ mx,
                         int C, int HW) {
  int p = blockIdx.x * blockDim.x + threadIdx.x;
  if (p >= HW) return;
  float m = -INFINITY;
  for (int c = 0; c < C; ++c) m = fmaxf(m, f[(size_t)c * HW + p]);
  mx[p] = m;
}

__device__ __forceinline__ float fval(const float* f, int y, int x, int H, int W) {
  return (y >= 0 && y < H && x >= 0 && x < W) ? f[(size_t)y * W + x] : 0.f;
}

__global__ void detect_k(const float* __restrict__ F, const float* __restrict__ mx,
                         float* __restrict__ out, int C, int H, int W) {
  int gid = blockIdx.x * blockDim.x + threadIdx.x;
  int total = C * H * W;
  if (gid >= total) return;
  int x = gid % W, y = (gid / W) % H, c = gid / (W * H);
  const float* f = F + (size_t)c * H * W;
  float c00 = fval(f, y - 1, x - 1, H, W), c01 = fval(f, y - 1, x, H, W), c02 = fval(f, y - 1, x + 1, H, W);
  float c10 = fval(f, y, x - 1, H, W),     c11 = f[(size_t)y * W + x],    c12 = fval(f, y, x + 1, H, W);
  float c20 = fval(f, y + 1, x - 1, H, W), c21 = fval(f, y + 1, x, H, W), c22 = fval(f, y + 1, x + 1, H, W);
  float dii = c01 + c21 - 2.f * c11;
  float djj = c10 + c12 - 2.f * c11;
  float dij = 0.25f * (c00 - c02 - c20 + c22);
  float det = dii * djj - dij * dij;
  float tr  = dii + djj;
  bool not_edge = ((tr * tr / det) <= 7.2f) && (det > 0.f);
  // features are post-ReLU (>=0): zero padding never exceeds the true max
  float lm = fmaxf(fmaxf(fmaxf(c00, c01), fmaxf(c02, c10)),
                   fmaxf(fmaxf(c11, c12), fmaxf(fmaxf(c20, c21), c22)));
  bool is_dw  = (c11 == mx[(size_t)y * W + x]);
  bool is_loc = (c11 == lm);
  out[gid] = (is_dw && is_loc && not_edge) ? 1.f : 0.f;
}

__global__ void localize_k(const float* __restrict__ F, float* __restrict__ disp,
                           int C, int H, int W) {
  int gid = blockIdx.x * blockDim.x + threadIdx.x;
  int total = C * H * W;
  if (gid >= total) return;
  int x = gid % W, y = (gid / W) % H, c = gid / (W * H);
  const float* f = F + (size_t)c * H * W;
  float c00 = fval(f, y - 1, x - 1, H, W), c01 = fval(f, y - 1, x, H, W), c02 = fval(f, y - 1, x + 1, H, W);
  float c10 = fval(f, y, x - 1, H, W),     c11 = f[(size_t)y * W + x],    c12 = fval(f, y, x + 1, H, W);
  float c20 = fval(f, y + 1, x - 1, H, W), c21 = fval(f, y + 1, x, H, W), c22 = fval(f, y + 1, x + 1, H, W);
  float dii = c01 + c21 - 2.f * c11;
  float djj = c10 + c12 - 2.f * c11;
  float dij = 0.25f * (c00 - c02 - c20 + c22);
  float det = dii * djj - dij * dij;
  float inv00 = djj / det, inv01 = -dij / det, inv11 = dii / det;
  float di = 0.5f * (c21 - c01);
  float dj = 0.5f * (c12 - c10);
  disp[gid]                 = -(inv00 * di + inv01 * dj);  // step_i
  disp[(size_t)total + gid] = -(inv01 * di + inv11 * dj);  // step_j
}

// ---------------------------------------------------------------------------
// Orchestration
// ---------------------------------------------------------------------------
extern "C" void kernel_launch(void* const* d_in, const int* in_sizes, int n_in,
                              void* d_out, int out_size, void* d_ws, size_t ws_size,
                              hipStream_t stream) {
  (void)in_sizes; (void)n_in; (void)out_size; (void)ws_size;
  const float* image = (const float*)d_in[0];
  const float* w[10]; const float* b[10];
  for (int i = 0; i < 10; ++i) { w[i] = (const float*)d_in[1 + 2 * i]; b[i] = (const float*)d_in[2 + 2 * i]; }

  const size_t BUF = 64ull * 768 * 768;              // largest layer (fp32 elems)
  float* buf0 = (float*)d_ws;
  float* buf1 = buf0 + BUF;
  bf16*  wt   = (bf16*)(buf1 + BUF);                 // up to 512*512*9 bf16
  float* mx   = (float*)(wt + 512ull * 512 * 9);     // 191*191 channel max

  auto g1 = [](size_t n) { return dim3((unsigned)((n + 255) / 256)); };

  // L1: 3 -> 64 @ 768, direct
  { size_t n = 64ull * 768 * 768;
    conv3x3_direct_c3<<<g1(n), 256, 0, stream>>>(image, w[0], b[0], buf0, 768, 768, 64); }

  auto convw = [&](const float* in, int li, float* outp,
                   int Cin, int Cout, int H, int W, int dil) {
    size_t nrep = (size_t)Cout * Cin * 9;
    repack_w_bf16<<<g1(nrep), 256, 0, stream>>>(w[li], wt, Cout, Cin);
    if (Cout >= 128) {
      dim3 g((W + 63) / 64, H, Cout / 128);
      conv3x3_wmma<2><<<g, 256, 0, stream>>>(in, wt, b[li], outp, Cin, Cout, H, W, dil);
    } else {
      dim3 g((W + 63) / 64, H, Cout / 64);
      conv3x3_wmma<1><<<g, 256, 0, stream>>>(in, wt, b[li], outp, Cin, Cout, H, W, dil);
    }
  };

  convw(buf0, 1, buf1, 64, 64, 768, 768, 1);
  { size_t n = 64ull * 384 * 384;
    maxpool2<<<g1(n), 256, 0, stream>>>(buf1, buf0, 64, 768, 768); }
  convw(buf0, 2, buf1, 64, 128, 384, 384, 1);
  convw(buf1, 3, buf0, 128, 128, 384, 384, 1);
  { size_t n = 128ull * 192 * 192;
    maxpool2<<<g1(n), 256, 0, stream>>>(buf0, buf1, 128, 384, 384); }
  convw(buf1, 4, buf0, 128, 256, 192, 192, 1);
  convw(buf0, 5, buf1, 256, 256, 192, 192, 1);
  convw(buf1, 6, buf0, 256, 256, 192, 192, 1);
  { size_t n = 256ull * 191 * 191;
    avgpool2_s1<<<g1(n), 256, 0, stream>>>(buf0, buf1, 256, 192, 192); }
  convw(buf1, 7, buf0, 256, 512, 191, 191, 2);
  convw(buf0, 8, buf1, 512, 512, 191, 191, 2);

  float* feat = (float*)d_out;                       // layer 10 writes features directly
  convw(buf1, 9, feat, 512, 512, 191, 191, 2);

  const int C = 512, H = 191, W = 191;
  const size_t FE = (size_t)C * H * W;
  chan_max<<<g1((size_t)H * W), 256, 0, stream>>>(feat, mx, C, H * W);
  detect_k<<<g1(FE), 256, 0, stream>>>(feat, mx, feat + FE, C, H, W);
  localize_k<<<g1(FE), 256, 0, stream>>>(feat, feat + 2 * FE, C, H, W);
}
